// Generator_84327387889752
// MI455X (gfx1250) — matrix operand order repaired and checked by reading, not compiled, not run on hardware
//
#include <hip/hip_runtime.h>
#include <stdint.h>
#include <stddef.h>

// ---------------- model dims ----------------
#define BB   16384
#define TT   10
#define FF   500
#define H1D  256
#define H2D  100
#define KX   512            // F padded to 512
#define ROWS 64             // batch rows per block
#define NTHREADS 512        // 16 waves (wave32)

// packed-weight element counts (bf16)
#define WXL_ELEMS (64*16*512)   // N=1024 (64 tiles) x K=512 (16 chunks)
#define WHL_ELEMS (64*8*512)    // N=1024 x K=256
#define WXG_ELEMS (21*8*512)    // N=336 (3 gates x 112) x K=256
#define WHG_ELEMS (21*4*512)    // N=336 x K=128

typedef __attribute__((ext_vector_type(16))) __bf16 v16bf;
typedef __attribute__((ext_vector_type(8)))  float  v8f;
typedef __attribute__((ext_vector_type(4)))  unsigned int u32x4;

// A-fragment (16x32 bf16, row-major LDS): lane half interleaves K by 8;
// two 16B chunks at k0 and k0+16.
__device__ __forceinline__ v16bf frag_gap(const __bf16* p) {
  union { v16bf v; u32x4 q[2]; } u;
  u.q[0] = *reinterpret_cast<const u32x4*>(p);
  u.q[1] = *reinterpret_cast<const u32x4*>(p + 16);
  return u.v;
}
// B-fragment from pre-swizzled (fragment-major) weights: 32B contiguous per lane.
__device__ __forceinline__ v16bf frag_lin(const __bf16* p) {
  union { v16bf v; u32x4 q[2]; } u;
  u.q[0] = *reinterpret_cast<const u32x4*>(p);
  u.q[1] = *reinterpret_cast<const u32x4*>(p + 8);
  return u.v;
}
__device__ __forceinline__ v8f wmma_bf(v16bf a, v16bf b, v8f c) {
  return __builtin_amdgcn_wmma_f32_16x16x32_bf16(false, a, false, b, (short)0, c,
                                                 false, false);
}
__device__ __forceinline__ float sigm(float x) { return 1.0f / (1.0f + __expf(-x)); }

// ---------------- weight pack: f32 row-major -> bf16 fragment-major ----------------
// dst layout: chunk = ntile*NKC + kc ; within chunk: lane L (0..31) * 16 + e (0..15)
//   element maps to src[k][n] with k = kc*32 + 8*(L/16) + (e%8) + 16*(e/8), n = ntile*16 + (L%16)
// gateReal!=0 remaps packed gate columns (gatePad per gate) back to src gate*gateReal+j.
__global__ void pack_b(const float* __restrict__ src, __bf16* __restrict__ dst,
                       int NT, int NKC, int realK, int srcLd, int gateReal, int gatePad) {
  int idx = blockIdx.x * blockDim.x + threadIdx.x;
  int total = NT * NKC * 512;
  if (idx >= total) return;
  int e     = idx & 15;
  int Ln    = (idx >> 4) & 31;
  int chunk = idx >> 9;
  int kc    = chunk % NKC;
  int ntile = chunk / NKC;
  int k = kc * 32 + ((Ln >> 4) << 3) + (e & 7) + ((e >> 3) << 4);
  int n = ntile * 16 + (Ln & 15);
  float v = 0.0f;
  bool ok = (k < realK);
  int nsrc = n;
  if (gateReal) {
    int g = n / gatePad, jj = n % gatePad;
    ok = ok && (jj < gateReal) && (g < 3);
    nsrc = g * gateReal + jj;
  }
  if (ok) v = src[(size_t)k * srcLd + nsrc];
  dst[idx] = (__bf16)v;
}

// ---------------- fully fused generator ----------------
__global__ __launch_bounds__(NTHREADS) void gen_fused(
    const float* __restrict__ noise,
    const __bf16* __restrict__ WxL, const __bf16* __restrict__ WhL,
    const float* __restrict__ bL,
    const __bf16* __restrict__ WxG, const __bf16* __restrict__ WhG,
    const float* __restrict__ bInG, const float* __restrict__ bRecG,
    const float* __restrict__ Wd, const float* __restrict__ bd,
    float* __restrict__ out)
{
  // LDS carve (row strides 16B aligned, bank-conflict-free for frag loads):
  //   x_lds  [64][520] bf16  = 66560 B   (x_t tile, K padded 512)
  //   h1_lds [64][264] bf16  = 33792 B   (LSTM hidden, K=256)
  //   h2_lds [64][136] bf16  = 17408 B   (GRU hidden, K padded 128)
  __shared__ __align__(16) unsigned char smem[117760];
  __bf16 (*x_lds)[520]  = reinterpret_cast<__bf16(*)[520]>(smem);
  __bf16 (*h1_lds)[264] = reinterpret_cast<__bf16(*)[264]>(smem + 66560);
  __bf16 (*h2_lds)[136] = reinterpret_cast<__bf16(*)[136]>(smem + 100352);

  const int tid  = threadIdx.x;
  const int w    = tid >> 5;      // wave 0..15
  const int L    = tid & 31;      // lane
  const int lh   = L >> 4;        // lane half
  const int lm   = L & 15;
  const int row0 = blockIdx.x * ROWS;

  // zero recurrent-state buffers (h1, h2) including padding columns
  {
    __bf16* hz = reinterpret_cast<__bf16*>(smem + 66560);
    for (int i = tid; i < (64 * 264 + 64 * 136); i += NTHREADS) hz[i] = (__bf16)0.0f;
  }

  // LSTM: wave w owns h-columns [w*16, w*16+16); per-lane gate biases
  const int ncol = w * 16 + lm;
  const float bl0 = bL[0 * H1D + ncol], bl1 = bL[1 * H1D + ncol];
  const float bl2 = bL[2 * H1D + ncol], bl3 = bL[3 * H1D + ncol];

  // Per-wave B base pointers (fragment-major packed weights).
  //   WxL chunk = (g*16+w)*16 + kc  -> gate stride 131072 elems, kc stride 512
  //   WhL chunk = (g*16+w)*8  + kc  -> gate stride  65536 elems, kc stride 512
  const __bf16* bxl0 = WxL + ((size_t)w * 16) * 512 + L * 16;
  const __bf16* bhl0 = WhL + ((size_t)w * 8)  * 512 + L * 16;

  // GRU: waves 0..13 -> (row-group rg 0..1) x (col-tile cgg 0..6)
  const int  rg  = w / 7;
  const int  cgg = w % 7;
  const int  j   = cgg * 16 + lm;      // padded gate-local column (0..111)
  const bool gv  = (w < 14);
  const bool jv  = gv && (j < H2D);
  const int  jc  = jv ? j : 0;
  const float bgz = jv ? (bInG[jc]           + bRecG[jc])           : 0.0f;
  const float bgr = jv ? (bInG[H2D + jc]     + bRecG[H2D + jc])     : 0.0f;
  const float bgi = jv ? bInG[2 * H2D + jc]  : 0.0f;
  const float bgh = jv ? bRecG[2 * H2D + jc] : 0.0f;
  //   WxG chunk = (g*7+cgg)*8 + kc -> gate stride 28672 elems, kc stride 512
  //   WhG chunk = (g*7+cgg)*4 + kc -> gate stride 14336 elems, kc stride 512
  const __bf16* bxg0 = WxG + ((size_t)cgg * 8) * 512 + L * 16;
  const __bf16* bhg0 = WhG + ((size_t)cgg * 4) * 512 + L * 16;

  const v8f zero = {};
  v8f cst[4] = {zero, zero, zero, zero};   // LSTM cell state (4 row tiles)
  v8f h2r[2] = {zero, zero};               // GRU hidden state (2 row tiles)

  __syncthreads();

  for (int t = 0; t < TT; ++t) {
    // ---- stage x_t into LDS as bf16 (float4 loads; 500-boundary is float4-aligned) ----
    {
      const float* nrowbase = noise + (size_t)row0 * (TT * FF) + t * FF;
      #pragma unroll 1
      for (int i = tid; i < ROWS * (KX / 4); i += NTHREADS) {
        int r = i >> 7, c4 = (i & 127) * 4;
        float4 v = make_float4(0.f, 0.f, 0.f, 0.f);
        if (c4 < FF) v = *reinterpret_cast<const float4*>(nrowbase + (size_t)r * (TT * FF) + c4);
        union { __bf16 b[4]; uint2 u; } pk;
        pk.b[0] = (__bf16)v.x; pk.b[1] = (__bf16)v.y;
        pk.b[2] = (__bf16)v.z; pk.b[3] = (__bf16)v.w;
        *reinterpret_cast<uint2*>(&x_lds[r][c4]) = pk.u;
      }
    }
    __syncthreads();

    // ---- LSTM: z = [x_t | h1] @ [WxL ; WhL]  (two loop-resident K loops, no unroll:
    //      live set = 128 acc + 32 A + 32 B VGPRs -> no spills) ----
    v8f acc[4][4];
    #pragma unroll
    for (int rt = 0; rt < 4; ++rt)
      #pragma unroll
      for (int g = 0; g < 4; ++g) acc[rt][g] = zero;

    {
      const __bf16* bp = bxl0;               // advances 512 elems (1KB) per chunk
      #pragma unroll 1
      for (int kc = 0; kc < 16; ++kc, bp += 512) {
        const int k0 = kc * 32 + 8 * lh;
        v16bf b[4];
        #pragma unroll
        for (int g = 0; g < 4; ++g) b[g] = frag_lin(bp + (size_t)g * 131072);
        v16bf a[4];
        #pragma unroll
        for (int rt = 0; rt < 4; ++rt) a[rt] = frag_gap(&x_lds[rt * 16 + lm][k0]);
        #pragma unroll
        for (int g = 0; g < 4; ++g)
          #pragma unroll
          for (int rt = 0; rt < 4; ++rt) acc[rt][g] = wmma_bf(a[rt], b[g], acc[rt][g]);
      }
    }
    {
      const __bf16* bp = bhl0;
      #pragma unroll 1
      for (int kc = 0; kc < 8; ++kc, bp += 512) {
        const int k0 = kc * 32 + 8 * lh;
        v16bf b[4];
        #pragma unroll
        for (int g = 0; g < 4; ++g) b[g] = frag_lin(bp + (size_t)g * 65536);
        v16bf a[4];
        #pragma unroll
        for (int rt = 0; rt < 4; ++rt) a[rt] = frag_gap(&h1_lds[rt * 16 + lm][k0]);
        #pragma unroll
        for (int g = 0; g < 4; ++g)
          #pragma unroll
          for (int rt = 0; rt < 4; ++rt) acc[rt][g] = wmma_bf(a[rt], b[g], acc[rt][g]);
      }
    }

    // gates (Keras order i,f,g,o; act=relu, rec_act=sigmoid)
    v8f hnew[4];
    #pragma unroll
    for (int rt = 0; rt < 4; ++rt) {
      #pragma unroll
      for (int e = 0; e < 8; ++e) {
        float ii = sigm(acc[rt][0][e] + bl0);
        float ff = sigm(acc[rt][1][e] + bl1);
        float gg = fmaxf(acc[rt][2][e] + bl2, 0.0f);
        float oo = sigm(acc[rt][3][e] + bl3);
        float c  = ff * cst[rt][e] + ii * gg;
        cst[rt][e] = c;
        hnew[rt][e] = oo * fmaxf(c, 0.0f);
      }
    }
    __syncthreads();   // everyone done reading h1_lds / x_lds for this step
    #pragma unroll
    for (int rt = 0; rt < 4; ++rt)
      #pragma unroll
      for (int e = 0; e < 8; ++e)     // D-tile: row = e + 8*lh, col = lm
        h1_lds[rt * 16 + e + 8 * lh][w * 16 + lm] = (__bf16)hnew[rt][e];
    __syncthreads();

    // ---- GRU step: K = 256 (h1) + 128 (h2 padded); N = 336 (3 gates x 112) ----
    if (gv) {   // wave-uniform branch: EXEC stays all-ones for WMMA
      v8f az[2] = {zero, zero}, ar[2] = {zero, zero};
      v8f ai[2] = {zero, zero}, ah[2] = {zero, zero};   // ai: input part of h~, ah: recurrent part
      {
        const __bf16* bp = bxg0;
        #pragma unroll 1
        for (int kc = 0; kc < 8; ++kc, bp += 512) {
          const int k0 = kc * 32 + 8 * lh;
          v16bf b[3];
          #pragma unroll
          for (int g = 0; g < 3; ++g) b[g] = frag_lin(bp + (size_t)g * 28672);
          v16bf a[2];
          #pragma unroll
          for (int rtl = 0; rtl < 2; ++rtl)
            a[rtl] = frag_gap(&h1_lds[(rg * 2 + rtl) * 16 + lm][k0]);
          #pragma unroll
          for (int rtl = 0; rtl < 2; ++rtl) {
            az[rtl] = wmma_bf(a[rtl], b[0], az[rtl]);
            ar[rtl] = wmma_bf(a[rtl], b[1], ar[rtl]);
            ai[rtl] = wmma_bf(a[rtl], b[2], ai[rtl]);
          }
        }
      }
      {
        const __bf16* bp = bhg0;
        #pragma unroll 1
        for (int kc = 0; kc < 4; ++kc, bp += 512) {
          const int k0 = kc * 32 + 8 * lh;
          v16bf b[3];
          #pragma unroll
          for (int g = 0; g < 3; ++g) b[g] = frag_lin(bp + (size_t)g * 14336);
          v16bf a[2];
          #pragma unroll
          for (int rtl = 0; rtl < 2; ++rtl)
            a[rtl] = frag_gap(&h2_lds[(rg * 2 + rtl) * 16 + lm][k0]);
          #pragma unroll
          for (int rtl = 0; rtl < 2; ++rtl) {
            az[rtl] = wmma_bf(a[rtl], b[0], az[rtl]);
            ar[rtl] = wmma_bf(a[rtl], b[1], ar[rtl]);
            ah[rtl] = wmma_bf(a[rtl], b[2], ah[rtl]);
          }
        }
      }
      #pragma unroll
      for (int rtl = 0; rtl < 2; ++rtl) {
        #pragma unroll
        for (int e = 0; e < 8; ++e) {
          float z  = sigm(az[rtl][e] + bgz);
          float r  = sigm(ar[rtl][e] + bgr);
          float hc = fmaxf(ai[rtl][e] + bgi + r * (ah[rtl][e] + bgh), 0.0f);
          float h  = z * h2r[rtl][e] + (1.0f - z) * hc;
          h2r[rtl][e] = jv ? h : 0.0f;   // keep padded columns exactly zero
        }
      }
    }
    __syncthreads();   // everyone done reading h2_lds
    if (gv) {
      #pragma unroll
      for (int rtl = 0; rtl < 2; ++rtl)
        #pragma unroll
        for (int e = 0; e < 8; ++e)
          h2_lds[(rg * 2 + rtl) * 16 + e + 8 * lh][cgg * 16 + lm] = (__bf16)h2r[rtl][e];
    }
    __syncthreads();
  }

  // ---- Dense head: out = relu(h2 @ Wd + bd), tiny -> plain VALU ----
  float (*hf)[112] = reinterpret_cast<float(*)[112]>(smem);  // alias x_lds region (x dead)
  if (gv) {
    #pragma unroll
    for (int rtl = 0; rtl < 2; ++rtl)
      #pragma unroll
      for (int e = 0; e < 8; ++e)
        hf[(rg * 2 + rtl) * 16 + e + 8 * lh][cgg * 16 + lm] = h2r[rtl][e];
  }
  __syncthreads();
  #pragma unroll 1
  for (int idx = tid; idx < ROWS * TT; idx += NTHREADS) {
    int r = idx / TT, jt = idx % TT;
    float s = bd[jt];
    for (int k = 0; k < H2D; ++k) s += hf[r][k] * Wd[k * TT + jt];
    out[(size_t)(row0 + r) * TT + jt] = fmaxf(s, 0.0f);
  }
}

// ---------------- launch ----------------
extern "C" void kernel_launch(void* const* d_in, const int* in_sizes, int n_in,
                              void* d_out, int out_size, void* d_ws, size_t ws_size,
                              hipStream_t stream) {
  const float* noise = (const float*)d_in[0];
  const float* WxLf  = (const float*)d_in[1];
  const float* WhLf  = (const float*)d_in[2];
  const float* bL    = (const float*)d_in[3];
  const float* WxGf  = (const float*)d_in[4];
  const float* WhGf  = (const float*)d_in[5];
  const float* bInG  = (const float*)d_in[6];
  const float* bRecG = (const float*)d_in[7];
  const float* Wd    = (const float*)d_in[8];
  const float* bd    = (const float*)d_in[9];
  float* out = (float*)d_out;

  char* ws = (char*)d_ws;
  __bf16* WxL = (__bf16*)(ws);
  __bf16* WhL = (__bf16*)(ws + 2ll * WXL_ELEMS);
  __bf16* WxG = (__bf16*)(ws + 2ll * (WXL_ELEMS + WHL_ELEMS));
  __bf16* WhG = (__bf16*)(ws + 2ll * (WXL_ELEMS + WHL_ELEMS + WXG_ELEMS));

  pack_b<<<(WXL_ELEMS + 255) / 256, 256, 0, stream>>>(WxLf, WxL, 64, 16, 500, 1024, 0, 1);
  pack_b<<<(WHL_ELEMS + 255) / 256, 256, 0, stream>>>(WhLf, WhL, 64, 8, 256, 1024, 0, 1);
  pack_b<<<(WXG_ELEMS + 255) / 256, 256, 0, stream>>>(WxGf, WxG, 21, 8, 256, 300, 100, 112);
  pack_b<<<(WHG_ELEMS + 255) / 256, 256, 0, stream>>>(WhGf, WhG, 21, 4, 100, 300, 100, 112);

  gen_fused<<<BB / ROWS, NTHREADS, 0, stream>>>(noise, WxL, WhL, bL, WxG, WhG,
                                                bInG, bRecG, Wd, bd, out);
}
